// SelfAttention_12610023981715
// MI455X (gfx1250) — matrix operand order repaired and checked by reading, not compiled, not run on hardware
//
#include <hip/hip_runtime.h>
#include <hip/hip_bf16.h>

// Shapes (fixed by the reference): B=4, C=256, Cqk=32, N=16*16*16=4096.
// d_in: x[4,256,4096] f32, Wq[32,256], bq[32], Wk[32,256], bk[32], Wv[256,256], bv[256]
// d_out: f32 [4,256,4096]
// Workspace layout (needs 10 MB):
//   qf16 [B,N,32]  f16 at ws + 0        (1 MB)
//   kf16 [B,N,32]  f16 at ws + 1 MB     (1 MB)
//   vt16 [B,256,N] f16 at ws + 2 MB     (8 MB)

typedef __attribute__((ext_vector_type(16))) _Float16 v16h;
typedef __attribute__((ext_vector_type(8)))  _Float16 v8h;
typedef __attribute__((ext_vector_type(8)))  float    v8f;
typedef __attribute__((ext_vector_type(4)))  float    v4f;

union V16U { v16h v; v8h h[2]; };

// A-operand (16x32 f16): lane = row (lane&15); lanes 0-15 hold K 0-7 & 16-23,
// lanes 16-31 hold K 8-15 & 24-31. p32 points at the row's 32 contiguous f16.
static __device__ inline v16h load_a16(const _Float16* p32, int lane) {
  const int koff = (lane >> 4) << 3;     // 0 or 8 elements
  V16U u;
  u.h[0] = *(const v8h*)(p32 + koff);
  u.h[1] = *(const v8h*)(p32 + 16 + koff);
  return u.v;
}

// B-operand (32x16 f16): lane = column (lane&15); lanes 0-15 hold K 0-7 & 16-23,
// lanes 16-31 hold K 8-15 & 24-31. p32 points at the
// column's 32 contiguous f16 (K-major).
static __device__ inline v16h load_b16(const _Float16* p32, int lane) {
  const int koff = (lane >> 4) << 3;
  V16U u;
  u.h[0] = *(const v8h*)(p32 + koff);
  u.h[1] = *(const v8h*)(p32 + 16 + koff);
  return u.v;
}

static __device__ inline v8f wmma_f16(v16h a, v16h b, v8f c) {
  return __builtin_amdgcn_wmma_f32_16x16x32_f16(false, a, false, b,
                                                (short)0, c, false, false);
}

// Max-reduce across a 16-lane group, pure VALU via DPP16 (no LDS, no waits):
//  xor1: quad_perm[1,0,3,2]=0xB1; xor2: quad_perm[2,3,0,1]=0x4E;
//  4-boundary: row_half_mirror=0x141 (4-groups already uniform);
//  8-boundary: row_mirror=0x140.
static __device__ inline float rowmax16(float v) {
  int t;
  t = __builtin_amdgcn_update_dpp(0, __float_as_int(v), 0xB1, 0xf, 0xf, true);
  v = fmaxf(v, __int_as_float(t));
  t = __builtin_amdgcn_update_dpp(0, __float_as_int(v), 0x4E, 0xf, 0xf, true);
  v = fmaxf(v, __int_as_float(t));
  t = __builtin_amdgcn_update_dpp(0, __float_as_int(v), 0x141, 0xf, 0xf, true);
  v = fmaxf(v, __int_as_float(t));
  t = __builtin_amdgcn_update_dpp(0, __float_as_int(v), 0x140, 0xf, 0xf, true);
  v = fmaxf(v, __int_as_float(t));
  return v;
}

// ---------------------------------------------------------------------------
// Kernel 1: q/k/v projections (1x1x1 convs). One block = one (b, 16-voxel)
// tile; x[:,tile] staged in LDS, each thread owns one output channel of v
// (threads 0-63 additionally own q/k rows).
// q,k stored [N,32] f16 row-major; v stored transposed [256,N] f16.
// ---------------------------------------------------------------------------
__global__ __launch_bounds__(256) void sattn_proj_kernel(
    const float* __restrict__ x,
    const float* __restrict__ Wq, const float* __restrict__ bq,
    const float* __restrict__ Wk, const float* __restrict__ bk,
    const float* __restrict__ Wv, const float* __restrict__ bv,
    _Float16* __restrict__ qf, _Float16* __restrict__ kf,
    _Float16* __restrict__ vt)
{
  constexpr int C = 256, Cqk = 32, N = 4096, NT = 16;
  __shared__ __align__(16) float xs[C * NT];   // xs[c*NT + j]

  const int t  = threadIdx.x;
  const int b  = blockIdx.x >> 8;              // 256 tiles per batch
  const int n0 = (blockIdx.x & 255) * NT;

  // Stage x tile: thread t loads channel t, 16 voxels (4x float4).
  const float* xp = x + ((size_t)b * C + t) * N + n0;
  v4f* xrow = (v4f*)&xs[t * NT];
  #pragma unroll
  for (int j = 0; j < NT / 4; ++j) xrow[j] = *(const v4f*)(xp + 4 * j);
  __syncthreads();

  // v projection: channel t over the 16 voxels. float4 weight loads.
  float accv[NT];
  #pragma unroll
  for (int j = 0; j < NT; ++j) accv[j] = 0.f;
  const v4f* wrow4 = (const v4f*)(Wv + t * C);
  for (int c4 = 0; c4 < C / 4; ++c4) {
    const v4f w = wrow4[c4];
    #pragma unroll
    for (int u = 0; u < 4; ++u)
      #pragma unroll
      for (int j = 0; j < NT; ++j)
        accv[j] = fmaf(w[u], xs[(c4 * 4 + u) * NT + j], accv[j]);
  }
  const float bvv = bv[t];
  _Float16* vdst = vt + ((size_t)b * C + t) * N + n0;
  #pragma unroll
  for (int j = 0; j < NT; ++j) vdst[j] = (_Float16)(accv[j] + bvv);

  // q (threads 0-31) / k (threads 32-63) projections.
  if (t < 64) {
    const int r = t & 31;
    const v4f* wp4 = (const v4f*)((((t < 32) ? Wq : Wk)) + r * C);
    const float bias = (t < 32) ? bq[r] : bk[r];
    float acc[NT];
    #pragma unroll
    for (int j = 0; j < NT; ++j) acc[j] = 0.f;
    for (int c4 = 0; c4 < C / 4; ++c4) {
      const v4f w = wp4[c4];
      #pragma unroll
      for (int u = 0; u < 4; ++u)
        #pragma unroll
        for (int j = 0; j < NT; ++j)
          acc[j] = fmaf(w[u], xs[(c4 * 4 + u) * NT + j], acc[j]);
    }
    _Float16* dst = ((t < 32) ? qf : kf) + ((size_t)b * N + n0) * Cqk + r;
    #pragma unroll
    for (int j = 0; j < NT; ++j) dst[j * Cqk] = (_Float16)(acc[j] + bias);
  }
}

// ---------------------------------------------------------------------------
// Kernel 2: flash attention + residual. One wave per 16-query tile.
// 64 keys per iteration: S = Q(16x32) @ K^T (4 WMMA), one online-softmax
// update (DPP-only row-max; row-sum from two extra ones-matrix WMMAs that
// ride the same rescale as the output accumulator), one P LDS round-trip,
// then 32 WMMAs accumulate out[16x256]. 38 WMMA / 64 keys; the per-block
// costs (136-mul rescale, max tree, dscnt wait) are paid half as often.
// ---------------------------------------------------------------------------
#define ATTN_WAVES 4

__global__ __launch_bounds__(32 * ATTN_WAVES) void sattn_flash_kernel(
    const float* __restrict__ x,
    const _Float16* __restrict__ qf, const _Float16* __restrict__ kf,
    const _Float16* __restrict__ vt, float* __restrict__ out)
{
  constexpr int C = 256, Cqk = 32, N = 4096;
  constexpr int CB = C / 16;                   // 16 channel blocks
  constexpr int KB = 64;                       // keys per iteration
  __shared__ __align__(64) _Float16 pl[ATTN_WAVES][16 * KB];

  const int lane = threadIdx.x & 31;
  const int wave = threadIdx.x >> 5;
  const int qt   = blockIdx.x * ATTN_WAVES + wave;   // 0..1023
  const int b    = qt >> 8;
  const int n0   = (qt & 255) << 4;
  const int rc   = lane & 15;                  // row for A, column for B
  const _Float16 h1 = (_Float16)1.0f;
  const v16h ones = {h1, h1, h1, h1, h1, h1, h1, h1,
                     h1, h1, h1, h1, h1, h1, h1, h1};

  // Q tile, A-operand resident for the whole kernel.
  const v16h qa = load_a16(qf + ((size_t)(b * N + n0) + rc) * Cqk, lane);

  const _Float16* kbase = kf + (size_t)b * N * Cqk;
  const _Float16* vbase = vt + (size_t)b * C * N;
  _Float16* myp = &pl[wave][0];

  v8f acc[CB] = {};                            // out[16 x 256] f32
  v8f acc_sum = {};                            // running softmax denominator
  float rowmax[8];
  #pragma unroll
  for (int j = 0; j < 8; ++j) rowmax[j] = -3.0e38f;

  for (int m0 = 0; m0 < N; m0 += KB) {
    // ---- scores: four 16-key columns, full K=Cqk=32 in one WMMA each ----
    v8f s[4];
    #pragma unroll
    for (int h = 0; h < 4; ++h) {
      const v16h kb = load_b16(kbase + (size_t)(m0 + 16 * h + rc) * Cqk, lane);
      v8f z = {};
      s[h] = wmma_f16(qa, kb, z);
    }

    // ---- online softmax (f32). D-layout: vgpr j = row j (lanes 0-15) /
    // row j+8 (lanes 16-31); the row's 16 columns span the 16-lane half. ----
    float fac[8];
    #pragma unroll
    for (int j = 0; j < 8; ++j) {
      float mv = fmaxf(fmaxf(s[0][j], s[1][j]), fmaxf(s[2][j], s[3][j]));
      mv = rowmax16(mv);                       // pure-VALU DPP reduce
      const float mn = fmaxf(rowmax[j], mv);
      fac[j] = __expf(rowmax[j] - mn);
      rowmax[j] = mn;
      // P (f16) into per-wave LDS tile, D-layout coordinates, 64-wide rows.
      const int r = ((lane < 16) ? j : (j + 8)) * KB;
      #pragma unroll
      for (int h = 0; h < 4; ++h)
        myp[r + 16 * h + rc] = (_Float16)__expf(s[h][j] - mn);
    }

    // Rescale running accumulators by exp(max_old - max_new).
    #pragma unroll
    for (int t = 0; t < CB; ++t)
      #pragma unroll
      for (int j = 0; j < 8; ++j) acc[t][j] *= fac[j];
    #pragma unroll
    for (int j = 0; j < 8; ++j) acc_sum[j] *= fac[j];

    // LDS round-trip: D-layout -> A-layout (two 16x32 A tiles per row).
    asm volatile("s_wait_dscnt 0" ::: "memory");
    const v16h pa0 = load_a16(myp + rc * KB, lane);
    const v16h pa1 = load_a16(myp + rc * KB + 32, lane);

    // Denominator for free: P @ ones (the two sum-WMMAs are separated by the
    // PV loop so the acc_sum RAW chain hides under independent WMMAs).
    acc_sum = wmma_f16(pa0, ones, acc_sum);

    // ---- out += P(16x64) @ V(64keys x 16ch), split by 32-key half ----
    #pragma unroll
    for (int t = 0; t < CB; ++t) {
      const v16h vb = load_b16(vbase + (size_t)(t * 16 + rc) * N + m0, lane);
      acc[t] = wmma_f16(pa0, vb, acc[t]);
    }

    acc_sum = wmma_f16(pa1, ones, acc_sum);

    #pragma unroll
    for (int t = 0; t < CB; ++t) {
      const v16h vb =
          load_b16(vbase + (size_t)(t * 16 + rc) * N + m0 + 32, lane);
      acc[t] = wmma_f16(pa1, vb, acc[t]);
    }
  }

  // ---- epilogue: normalize, residual add, store ----
  float rinv[8];
  #pragma unroll
  for (int j = 0; j < 8; ++j) rinv[j] = 1.0f / acc_sum[j];

  #pragma unroll
  for (int t = 0; t < CB; ++t) {
    #pragma unroll
    for (int j = 0; j < 8; ++j) {
      const int row = (lane < 16) ? j : (j + 8);        // query in tile
      const int ch  = t * 16 + rc;                      // channel
      const size_t idx = ((size_t)b * C + ch) * (size_t)N + (size_t)(n0 + row);
      out[idx] = x[idx] + acc[t][j] * rinv[j];
    }
  }
}

extern "C" void kernel_launch(void* const* d_in, const int* in_sizes, int n_in,
                              void* d_out, int out_size, void* d_ws, size_t ws_size,
                              hipStream_t stream) {
  (void)in_sizes; (void)n_in; (void)out_size; (void)ws_size;
  const float* x  = (const float*)d_in[0];
  const float* Wq = (const float*)d_in[1];
  const float* bq = (const float*)d_in[2];
  const float* Wk = (const float*)d_in[3];
  const float* bk = (const float*)d_in[4];
  const float* Wv = (const float*)d_in[5];
  const float* bv = (const float*)d_in[6];
  float* out = (float*)d_out;

  char* ws = (char*)d_ws;
  _Float16* qf = (_Float16*)(ws);
  _Float16* kf = (_Float16*)(ws + (1u << 20));
  _Float16* vt = (_Float16*)(ws + (2u << 20));

  // B * (N/16) = 1024 tiles.
  sattn_proj_kernel<<<dim3(1024), dim3(256), 0, stream>>>(
      x, Wq, bq, Wk, bk, Wv, bv, qf, kf, vt);
  sattn_flash_kernel<<<dim3(1024 / ATTN_WAVES), dim3(32 * ATTN_WAVES), 0, stream>>>(
      x, qf, kf, vt, out);
}
